// NeoBottleNeck_82274393522961
// MI455X (gfx1250) — matrix operand order repaired and checked by reading, not compiled, not run on hardware
//
#include <hip/hip_runtime.h>
#include <math.h>

typedef __attribute__((ext_vector_type(8)))  float    v8f;
typedef __attribute__((ext_vector_type(8)))  _Float16 v8h;
typedef __attribute__((ext_vector_type(16))) _Float16 v16h;

#define CC    256
#define C1    128
#define HH    56
#define WW    56
#define CI    1024
#define YS    264   // y tile row stride (halfs): 16B-aligned rows, bank-staggered
#define HS    136   // hidden chunk row stride (halfs)
#define OSTR  260   // epilogue float staging stride

// Pre-split f16 hi/lo weights (1024*256 each = 512KB; 2MB total, L2-resident)
__device__ _Float16 g_w1hi[CI * CC];
__device__ _Float16 g_w1lo[CI * CC];
__device__ _Float16 g_w2hi[CC * CI];
__device__ _Float16 g_w2lo[CC * CI];

__global__ void split_weights(const float* __restrict__ w1, const float* __restrict__ w2) {
    int i = blockIdx.x * 256 + threadIdx.x;          // grid covers 262144 elements
    float a = w1[i];
    _Float16 ah = (_Float16)a;
    g_w1hi[i] = ah;
    g_w1lo[i] = (_Float16)(a - (float)ah);
    float b = w2[i];
    _Float16 bh = (_Float16)b;
    g_w2hi[i] = bh;
    g_w2lo[i] = (_Float16)(b - (float)bh);
}

// Branchless exact-GELU: 0.5*x*(1+erf(x/sqrt2)), erf via A&S 7.1.26 (|err|<1.5e-7)
__device__ __forceinline__ float gelu_exact(float x) {
    float z = 0.70710678118654752f * x;
    float s = fabsf(z);
    float t = __builtin_amdgcn_rcpf(1.0f + 0.3275911f * s);
    float p = t * (0.254829592f +
              t * (-0.284496736f +
              t * (1.421413741f +
              t * (-1.453152027f +
              t * 1.061405429f))));
    float e  = __expf(-s * s);
    float er = copysignf(1.0f - p * e, z);
    return 0.5f * x * (1.0f + er);
}

// A fragment (16x32 f16): lane m = lane&15; half-sel hi: contiguous K run [hi*16, hi*16+16)
__device__ __forceinline__ v16h load_a_frag(const _Float16* rowp, int k0, int hi) {
    union { v16h v; v8h h[2]; } u;
    u.h[0] = *(const v8h*)(rowp + k0 + hi * 16);
    u.h[1] = *(const v8h*)(rowp + k0 + hi * 16 + 8);
    return u.v;
}
// B fragment (32x16 f16): lane n holds contiguous K run of 16 at khalf
__device__ __forceinline__ v16h load_b_frag(const _Float16* rowp, int khalf) {
    union { v16h v; v8h h[2]; } u;
    u.h[0] = *(const v8h*)(rowp + khalf);
    u.h[1] = *(const v8h*)(rowp + khalf + 8);
    return u.v;
}

#define WMMA(A, B, Cacc) __builtin_amdgcn_wmma_f32_16x16x32_f16(false, (A), false, (B), (short)0, (Cacc), false, false)

__global__ __launch_bounds__(256, 2)
void neocell_mlp_fused(const float* __restrict__ x,
                       const float* __restrict__ wa1, const float* __restrict__ wb1,
                       const float* __restrict__ wa2, const float* __restrict__ wb2,
                       const float* __restrict__ bnw, const float* __restrict__ bnb,
                       const float* __restrict__ bnm, const float* __restrict__ bnv,
                       float* __restrict__ out)
{
    // LDS carve: yhi|ylo (split activations) + hhi|hlo (split hidden chunk);
    // the y region is reused as fp32 output staging in the epilogue.
    __shared__ __align__(16) unsigned char smem[2 * 16 * YS * 2 + 2 * 16 * HS * 2];
    _Float16* yhi = (_Float16*)smem;                 // 16*264 halfs
    _Float16* ylo = yhi + 16 * YS;
    _Float16* hhi = ylo + 16 * YS;                   // 16*136 halfs
    _Float16* hlo = hhi + 16 * HS;
    float*    ost = (float*)smem;                    // 16*260 floats (fits in y region)

    const int b  = blockIdx.z;
    const int h0 = blockIdx.y * 4;
    const int w0 = blockIdx.x * 4;
    const int c  = threadIdx.x;

    // ---------------- Prologue: NeoCell + BatchNorm -> split-f16 LDS ----------------
    float xp[16];
    {
        const float* xc = x + (((size_t)(b * CC + c) * HH + h0) * WW + w0);
        #pragma unroll
        for (int i = 0; i < 4; ++i) {
            float4 v = *(const float4*)(xc + i * WW);
            xp[i*4+0] = v.x; xp[i*4+1] = v.y; xp[i*4+2] = v.z; xp[i*4+3] = v.w;
        }
        float yv[16];
        if (c < C1) {
            float A0 = wa1[c*4+0], A1 = wa1[c*4+1], A2 = wa1[c*4+2], A3 = wa1[c*4+3];
            float B0 = wb1[c*4+0], B1 = wb1[c*4+1], B2 = wb1[c*4+2], B3 = wb1[c*4+3];
            #pragma unroll
            for (int bi = 0; bi < 2; ++bi)
                #pragma unroll
                for (int bj = 0; bj < 2; ++bj) {
                    float x00 = xp[(2*bi+0)*4 + 2*bj+0], x01 = xp[(2*bi+0)*4 + 2*bj+1];
                    float x10 = xp[(2*bi+1)*4 + 2*bj+0], x11 = xp[(2*bi+1)*4 + 2*bj+1];
                    float t00 = A0*x00 + A1*x10, t01 = A0*x01 + A1*x11;
                    float t10 = A2*x00 + A3*x10, t11 = A2*x01 + A3*x11;
                    yv[(2*bi+0)*4 + 2*bj+0] = t00*B0 + t01*B2;
                    yv[(2*bi+0)*4 + 2*bj+1] = t00*B1 + t01*B3;
                    yv[(2*bi+1)*4 + 2*bj+0] = t10*B0 + t11*B2;
                    yv[(2*bi+1)*4 + 2*bj+1] = t10*B1 + t11*B3;
                }
        } else {
            const float* A  = wa2 + (size_t)(c - C1) * 16;
            const float* Bm = wb2 + (size_t)(c - C1) * 16;
            float t[16];
            #pragma unroll
            for (int p = 0; p < 4; ++p)
                #pragma unroll
                for (int j = 0; j < 4; ++j)
                    t[p*4+j] = A[p*4+0]*xp[0*4+j] + A[p*4+1]*xp[1*4+j]
                             + A[p*4+2]*xp[2*4+j] + A[p*4+3]*xp[3*4+j];
            #pragma unroll
            for (int p = 0; p < 4; ++p)
                #pragma unroll
                for (int q = 0; q < 4; ++q)
                    yv[p*4+q] = t[p*4+0]*Bm[0*4+q] + t[p*4+1]*Bm[1*4+q]
                              + t[p*4+2]*Bm[2*4+q] + t[p*4+3]*Bm[3*4+q];
        }
        float inv = rsqrtf(bnv[c] + 1e-5f);
        float sc  = bnw[c] * inv;
        float sh  = bnb[c] - bnm[c] * sc;
        #pragma unroll
        for (int m = 0; m < 16; ++m) {
            float v = yv[m] * sc + sh;
            _Float16 vh = (_Float16)v;
            yhi[m * YS + c] = vh;
            ylo[m * YS + c] = (_Float16)(v - (float)vh);
        }
    }
    __syncthreads();

    // ---------------- Fused MLP: split-f16 WMMA (hi*hi + hi*lo + lo*hi) ----------------
    const int lane = threadIdx.x & 31;
    const int wv   = threadIdx.x >> 5;
    const int lm   = lane & 15;
    const int hi   = lane >> 4;

    v8f o0 = {};   // out channels [wv*32,    wv*32+16)
    v8f o1 = {};   // out channels [wv*32+16, wv*32+32)

    const _Float16* a1h = yhi + lm * YS;
    const _Float16* a1l = ylo + lm * YS;
    const _Float16* a2h = hhi + lm * HS;
    const _Float16* a2l = hlo + lm * HS;
    const _Float16* b2h0 = g_w2hi + (size_t)(wv * 32 + lm) * CI;
    const _Float16* b2l0 = g_w2lo + (size_t)(wv * 32 + lm) * CI;
    const _Float16* b2h1 = b2h0 + (size_t)16 * CI;
    const _Float16* b2l1 = b2l0 + (size_t)16 * CI;

    for (int j = 0; j < 8; ++j) {
        const int nH = j * 128;
        // --- fc1: hidden tile columns [nH + wv*16, +16) ---
        v8f acc = {};
        const _Float16* b1h = g_w1hi + (size_t)(nH + wv * 16 + lm) * CC;
        const _Float16* b1l = g_w1lo + (size_t)(nH + wv * 16 + lm) * CC;
        #pragma unroll
        for (int k0 = 0; k0 < CC; k0 += 32) {
            v16h ah = load_a_frag(a1h, k0, hi);
            v16h al = load_a_frag(a1l, k0, hi);
            int  kb = k0 + hi * 16;
            v16h bh = load_b_frag(b1h, kb);
            v16h bl = load_b_frag(b1l, kb);
            acc = WMMA(ah, bh, acc);
            acc = WMMA(ah, bl, acc);
            acc = WMMA(al, bh, acc);
        }
        // GELU + split-f16 stage of hidden chunk
        #pragma unroll
        for (int r = 0; r < 8; ++r) {
            float g = gelu_exact(acc[r]);
            _Float16 gh = (_Float16)g;
            int row = r + hi * 8;
            hhi[row * HS + wv * 16 + lm] = gh;
            hlo[row * HS + wv * 16 + lm] = (_Float16)(g - (float)gh);
        }
        __syncthreads();

        // --- fc2 partial over this hidden chunk ---
        #pragma unroll
        for (int k0 = 0; k0 < 128; k0 += 32) {
            v16h ah = load_a_frag(a2h, k0, hi);
            v16h al = load_a_frag(a2l, k0, hi);
            int  kb = nH + k0 + hi * 16;
            v16h c0h = load_b_frag(b2h0, kb);
            v16h c0l = load_b_frag(b2l0, kb);
            v16h c1h = load_b_frag(b2h1, kb);
            v16h c1l = load_b_frag(b2l1, kb);
            o0 = WMMA(ah, c0h, o0);
            o0 = WMMA(ah, c0l, o0);
            o0 = WMMA(al, c0h, o0);
            o1 = WMMA(ah, c1h, o1);
            o1 = WMMA(ah, c1l, o1);
            o1 = WMMA(al, c1h, o1);
        }
        __syncthreads();
    }

    // ---------------- Epilogue: stage fp32 through LDS, residual, float4 stores ----
    #pragma unroll
    for (int r = 0; r < 8; ++r) {
        int m = r + hi * 8;
        ost[m * OSTR + wv * 32 + lm]      = o0[r];
        ost[m * OSTR + wv * 32 + 16 + lm] = o1[r];
    }
    __syncthreads();

    float* op = out + (((size_t)(b * CC + c) * HH + h0) * WW + w0);
    #pragma unroll
    for (int dy = 0; dy < 4; ++dy) {
        float4 rr;
        rr.x = ost[(dy*4+0) * OSTR + c] + xp[dy*4+0];
        rr.y = ost[(dy*4+1) * OSTR + c] + xp[dy*4+1];
        rr.z = ost[(dy*4+2) * OSTR + c] + xp[dy*4+2];
        rr.w = ost[(dy*4+3) * OSTR + c] + xp[dy*4+3];
        *(float4*)(op + dy * WW) = rr;
    }
}

extern "C" void kernel_launch(void* const* d_in, const int* in_sizes, int n_in,
                              void* d_out, int out_size, void* d_ws, size_t ws_size,
                              hipStream_t stream) {
    const float* x     = (const float*)d_in[0];
    const float* wa1   = (const float*)d_in[1];
    const float* wb1   = (const float*)d_in[2];
    const float* wa2   = (const float*)d_in[3];
    const float* wb2   = (const float*)d_in[4];
    const float* bnw   = (const float*)d_in[5];
    const float* bnb   = (const float*)d_in[6];
    const float* bnm   = (const float*)d_in[7];
    const float* bnv   = (const float*)d_in[8];
    const float* fc1_w = (const float*)d_in[9];
    const float* fc2_w = (const float*)d_in[10];

    split_weights<<<(CI * CC) / 256, 256, 0, stream>>>(fc1_w, fc2_w);

    dim3 grid(WW / 4, HH / 4, 32);   // 14 x 14 x 32 workgroups
    neocell_mlp_fused<<<grid, 256, 0, stream>>>(x, wa1, wb1, wa2, wb2,
                                                bnw, bnb, bnm, bnv,
                                                (float*)d_out);
}
